// BiMambaBlock_21509196218761
// MI455X (gfx1250) — matrix operand order repaired and checked by reading, not compiled, not run on hardware
//
#include <hip/hip_runtime.h>
#include <hip/hip_bf16.h>
#include <math.h>

#define DIM        1024
#define D_STATE    16
#define D_CONV     4
#define D_INNER    2048
#define NHEADS     32
#define HEADDIM    64
#define CONV_DIM   2080
#define D_IN_PROJ  4160
#define MLP_HID    4096
#define B_SZ       4
#define SEQ_L      1024
#define M_TOK      (B_SZ * SEQ_L)

#if __has_builtin(__builtin_amdgcn_global_load_async_to_lds_b128) && \
    __has_builtin(__builtin_amdgcn_s_wait_asynccnt)
#define HAS_ASYNC_LDS 1
#else
#define HAS_ASYNC_LDS 0
#endif

typedef unsigned short u16;
typedef __attribute__((ext_vector_type(16))) __bf16 v16bf;
typedef __attribute__((ext_vector_type(8)))  __bf16 bfx8;
typedef __attribute__((ext_vector_type(8)))  float  v8f;
typedef __attribute__((ext_vector_type(4)))  int    v4i;

typedef __attribute__((address_space(1))) v4i* gv4i_ptr;  // global int4*
typedef __attribute__((address_space(3))) v4i* lv4i_ptr;  // LDS int4*

__device__ __forceinline__ u16 f2bf(float f) {
  unsigned u = __float_as_uint(f);
  return (u16)((u + 0x7FFFu + ((u >> 16) & 1u)) >> 16);   // RNE
}
__device__ __forceinline__ float bf2f(u16 h) {
  return __uint_as_float(((unsigned)h) << 16);
}
__device__ __forceinline__ float siluf(float x) { return x / (1.f + __expf(-x)); }

// ---------------------------------------------------------------------------
// bf16 WMMA GEMM:  C[m,n] = sum_k A[m,k] * W[n,k]   (both K-contiguous)
// 256 threads = 8 waves, 128x128 block tile, K-step 32.
// Wave grid 4(M) x 2(N): each wave owns 32 rows x 64 cols -> 2 A-frags,
// 4 B-frags, 8 WMMAs per K-step.
// A tile: async global->LDS copy (GLOBAL_LOAD_ASYNC_TO_LDS_B128, ASYNCcnt),
// double-buffered so tile k+1 copies while tile k computes. W tile: register-
// staged + transpose-on-stage scatter into k-major sB so the B fragment
// (lane = n, 16 k-values per lane) reads one sB column, lanes on consecutive n.
// Epilogue: optional bias / exact-GELU / residual; fp32 or bf16 output with
// arbitrary ldc + column offset (writes directly into concat halves).
// ---------------------------------------------------------------------------
template <bool OUT_BF16, bool GELU>
__global__ __launch_bounds__(256) void gemm_wmma_bf16(
    const u16* __restrict__ A, const u16* __restrict__ W,
    float* __restrict__ Cf, u16* __restrict__ Cb,
    const float* __restrict__ bias, const float* __restrict__ resid,
    int N, int K, int ldc, int cc0) {
#if HAS_ASYNC_LDS
  __shared__ __attribute__((aligned(32))) u16 sA[2][128][40];  // double buffer
#else
  __shared__ __attribute__((aligned(32))) u16 sA[1][128][40];
#endif
  __shared__ __attribute__((aligned(32))) u16 sB[32][144];     // k-major, 288B rows

  const int tid  = threadIdx.x;
  const int lane = tid & 31;
  const int wave = tid >> 5;
  const int half = lane >> 4;
  const int l16  = lane & 15;
  const int wave_m = wave >> 1;   // 0..3 -> 32-row strip
  const int wave_n = wave & 1;    // 0..1 -> 64-col strip
  const int bn = blockIdx.x * 128;
  const int bm = blockIdx.y * 128;

  v8f acc[2][4] = {};   // fp32 accumulators: 2 M-tiles x 4 N-tiles

  const int srow = tid >> 2;        // 0..63
  const int scol = (tid & 3) * 8;   // 0,8,16,24 (k offset, 8 bf16 = 16B)

  // ---- W pipeline registers ----
  uint4 rw[2];
  auto load_W = [&](int k0) {
#pragma unroll
    for (int it = 0; it < 2; ++it) {
      int gn = bn + srow + it * 64;
      if (gn < N) {
        const u16* gw = W + (size_t)gn * K + (k0 + scol);
        rw[it] = *(const uint4*)gw;
        if (k0 + 32 < K) __builtin_prefetch(gw + 32, 0, 1);  // global_prefetch_b8
      } else {
        rw[it] = make_uint4(0u, 0u, 0u, 0u);
      }
    }
  };

#if HAS_ASYNC_LDS
  auto async_A = [&](int k0, int buf) {
#pragma unroll
    for (int it = 0; it < 2; ++it) {
      int row = srow + it * 64;
      const u16* ga = A + (size_t)(bm + row) * K + (k0 + scol);
      __builtin_amdgcn_global_load_async_to_lds_b128(
          (gv4i_ptr)(void*)const_cast<u16*>(ga),
          (lv4i_ptr)(void*)&sA[buf][row][scol],
          0, 0);
    }
  };
  async_A(0, 0);
#else
  uint4 ra[2];
  auto load_A = [&](int k0) {
#pragma unroll
    for (int it = 0; it < 2; ++it) {
      const u16* ga = A + (size_t)(bm + srow + it * 64) * K + (k0 + scol);
      ra[it] = *(const uint4*)ga;
      if (k0 + 32 < K) __builtin_prefetch(ga + 32, 0, 1);
    }
  };
  load_A(0);
#endif
  load_W(0);

  int cur = 0;
  for (int k0 = 0; k0 < K; k0 += 32) {
    __syncthreads();   // previous compute finished reading sB (and old sA buffer)
    // stage W (transpose-on-stage into k-major sB)
#pragma unroll
    for (int it = 0; it < 2; ++it) {
      int row = srow + it * 64;
      union { uint4 q; u16 s[8]; } wv;
      wv.q = rw[it];
#pragma unroll
      for (int j = 0; j < 8; ++j) sB[scol + j][row] = wv.s[j];
#if !HAS_ASYNC_LDS
      *(uint4*)&sA[0][row][scol] = ra[it];
#endif
    }
    const bool more = (k0 + 32 < K);
    if (more) {
      load_W(k0 + 32);            // next W tile into regs, no wait yet
#if HAS_ASYNC_LDS
      async_A(k0 + 32, cur ^ 1);  // next A tile straight into the other buffer
#else
      load_A(k0 + 32);
#endif
    }
#if HAS_ASYNC_LDS
    // current A tile (issued last iteration) must have landed; the 2 newest
    // async ops (next tile) stay in flight and overlap this tile's WMMAs.
    if (more) __builtin_amdgcn_s_wait_asynccnt(2);
    else      __builtin_amdgcn_s_wait_asynccnt(0);
#endif
    __syncthreads();

    // A fragments: lane<16 -> row m=l16, k {0..7,16..23}; lane>=16 -> k {8..15,24..31}
    union Frag { v16bf v; struct { bfx8 lo, hi; } p; };
    Frag fa[2];
#pragma unroll
    for (int i = 0; i < 2; ++i) {
      const u16* pa = &sA[cur][wave_m * 32 + i * 16 + l16][half * 8];
      fa[i].p.lo = *(const bfx8*)pa;
      fa[i].p.hi = *(const bfx8*)(pa + 16);
    }
#pragma unroll
    for (int j = 0; j < 4; ++j) {
      // B fragment: lane -> col n=l16, elements = k in same order as A fragment
      v16bf fb;
#pragma unroll
      for (int s = 0; s < 16; ++s)
        fb[s] = *(const __bf16*)&sB[(s < 8) ? (half * 8 + s) : (16 + half * 8 + s - 8)][wave_n * 64 + j * 16 + l16];
#pragma unroll
      for (int i = 0; i < 2; ++i) {
        acc[i][j] = __builtin_amdgcn_wmma_f32_16x16x32_bf16(
            false, fa[i].v, false, fb, (short)0, acc[i][j], false, false);
      }
    }
#if HAS_ASYNC_LDS
    cur ^= 1;
#endif
  }

  // C/D layout: lanes 0-15: vgpr ii -> M=ii, N=lane; lanes 16-31: M=8+ii, N=lane-16
#pragma unroll
  for (int j = 0; j < 4; ++j) {
    int n = bn + wave_n * 64 + j * 16 + l16;
    if (n >= N) continue;
    float bv = bias ? bias[n] : 0.f;
#pragma unroll
    for (int i = 0; i < 2; ++i) {
#pragma unroll
      for (int ii = 0; ii < 8; ++ii) {
        int m = bm + wave_m * 32 + i * 16 + half * 8 + ii;
        float v = acc[i][j][ii] + bv;
        if (GELU) v = 0.5f * v * (1.f + erff(v * 0.70710678118f));
        if (resid) v += resid[(size_t)m * N + n];
        size_t off = (size_t)m * ldc + cc0 + n;
        if (OUT_BF16) Cb[off] = f2bf(v); else Cf[off] = v;
      }
    }
  }
}

// --------------------------------------------------------------------------
// LayerNorm over DIM=1024 -> bf16 (GEMM operand feed)
// --------------------------------------------------------------------------
__global__ __launch_bounds__(256) void layernorm_to_bf16(
    const float* __restrict__ x, const float* __restrict__ w,
    const float* __restrict__ b, u16* __restrict__ out) {
  __shared__ float r1[256], r2[256];
  const int row = blockIdx.x;
  const int tid = threadIdx.x;
  const float* xr = x + (size_t)row * DIM;
  float v[4];
  float s1 = 0.f, s2 = 0.f;
#pragma unroll
  for (int i = 0; i < 4; ++i) {
    v[i] = xr[tid * 4 + i];
    s1 += v[i]; s2 += v[i] * v[i];
  }
  r1[tid] = s1; r2[tid] = s2;
  __syncthreads();
  for (int st = 128; st > 0; st >>= 1) {
    if (tid < st) { r1[tid] += r1[tid + st]; r2[tid] += r2[tid + st]; }
    __syncthreads();
  }
  float mu   = r1[0] * (1.f / DIM);
  float var  = r2[0] * (1.f / DIM) - mu * mu;
  float rstd = rsqrtf(var + 1e-5f);
#pragma unroll
  for (int i = 0; i < 4; ++i) {
    int c = tid * 4 + i;
    out[(size_t)row * DIM + c] = f2bf((v[i] - mu) * rstd * w[c] + b[c]);
  }
}

// --------------------------------------------------------------------------
// Causal depthwise conv (K=4) + SiLU over the xBC slice of zxbcdt.
// dir=1 runs in flipped time; output kept in scan order.
// --------------------------------------------------------------------------
__global__ __launch_bounds__(256) void conv_silu(
    const u16* __restrict__ zx, const float* __restrict__ cw,
    const float* __restrict__ cb, float* __restrict__ out, int dir) {
  int idx = blockIdx.x * 256 + threadIdx.x;
  if (idx >= M_TOK * CONV_DIM) return;
  int c  = idx % CONV_DIM;
  int r  = idx / CONV_DIM;        // scan-order row (b*L + s)
  int s  = r & (SEQ_L - 1);
  int bb = r >> 10;
  float acc = cb[c];
#pragma unroll
  for (int k = 0; k < D_CONV; ++k) {
    int j = s - (D_CONV - 1) + k;
    if (j >= 0) {
      int l = dir ? (SEQ_L - 1 - j) : j;
      acc += bf2f(zx[(size_t)(bb * SEQ_L + l) * D_IN_PROJ + D_INNER + c]) * cw[c * D_CONV + k];
    }
  }
  out[(size_t)r * CONV_DIM + c] = siluf(acc);
}

// --------------------------------------------------------------------------
// Selective scan: one block per (batch, head); thread t owns state[p=t/4, 4 n's].
// y contraction over N reduced across 4 wave32 lanes via shfl_xor.
// --------------------------------------------------------------------------
__global__ __launch_bounds__(256) void ssm_scan(
    const u16* __restrict__ zx, const float* __restrict__ xbc,
    const float* __restrict__ dt_bias, const float* __restrict__ A_log,
    const float* __restrict__ Dp, float* __restrict__ y, int dir) {
  const int bh = blockIdx.x;
  const int bb = bh >> 5, hh = bh & 31;
  const int tid = threadIdx.x;
  const int p = tid >> 2, nq = (tid & 3) * 4;
  float st[4] = {0.f, 0.f, 0.f, 0.f};
  const float Am  = -__expf(A_log[hh]);
  const float Dh  = Dp[hh];
  const float dtb = dt_bias[hh];
  for (int s = 0; s < SEQ_L; ++s) {
    int l = dir ? (SEQ_L - 1 - s) : s;
    size_t rowO = (size_t)(bb * SEQ_L + l) * D_IN_PROJ;  // original order (dt lives here)
    size_t rowS = (size_t)(bb * SEQ_L + s) * CONV_DIM;   // scan order (conv output)
    float dtr = bf2f(zx[rowO + D_INNER + CONV_DIM + hh]) + dtb;
    float dtv = (dtr > 20.f) ? dtr : log1pf(__expf(dtr));  // softplus
    float dA  = __expf(Am * dtv);
    float xv  = xbc[rowS + hh * HEADDIM + p];
    float coef = dtv * xv;
    float yp = 0.f;
#pragma unroll
    for (int j = 0; j < 4; ++j) {
      float Bv = xbc[rowS + D_INNER + nq + j];
      float Cv = xbc[rowS + D_INNER + D_STATE + nq + j];
      st[j] = dA * st[j] + coef * Bv;
      yp += st[j] * Cv;
    }
    yp += __shfl_xor(yp, 1, 32);
    yp += __shfl_xor(yp, 2, 32);
    if ((tid & 3) == 0)
      y[(size_t)(bb * SEQ_L + s) * D_INNER + hh * HEADDIM + p] = yp + Dh * xv;
  }
}

// --------------------------------------------------------------------------
// Gated RMSNorm: out = rmsnorm(y * silu(z)) * w, written un-flipped, bf16.
// --------------------------------------------------------------------------
__global__ __launch_bounds__(256) void gated_rmsnorm(
    const float* __restrict__ y, const u16* __restrict__ zx,
    const float* __restrict__ nw, u16* __restrict__ out, int dir) {
  __shared__ float red[256];
  const int r  = blockIdx.x;                 // scan-order row
  const int bb = r >> 10, s = r & (SEQ_L - 1);
  const int l  = dir ? (SEQ_L - 1 - s) : s;  // original position
  const int tid = threadIdx.x;
  const size_t rowO = (size_t)(bb * SEQ_L + l);
  float g[8];
  float ss = 0.f;
#pragma unroll
  for (int i = 0; i < 8; ++i) {
    int c = tid * 8 + i;
    float zv = bf2f(zx[rowO * D_IN_PROJ + c]);
    float gv = y[(size_t)r * D_INNER + c] * siluf(zv);
    g[i] = gv; ss += gv * gv;
  }
  red[tid] = ss;
  __syncthreads();
  for (int st = 128; st > 0; st >>= 1) {
    if (tid < st) red[tid] += red[tid + st];
    __syncthreads();
  }
  float rms = rsqrtf(red[0] * (1.f / D_INNER) + 1e-5f);
#pragma unroll
  for (int i = 0; i < 8; ++i) {
    int c = tid * 8 + i;
    out[rowO * D_INNER + c] = f2bf(g[i] * rms * nw[c]);
  }
}

__global__ void cvt_bf16(const float* __restrict__ in, u16* __restrict__ out, int n) {
  int i = blockIdx.x * 256 + threadIdx.x;
  if (i < n) out[i] = f2bf(in[i]);
}

// ---------------------------------------------------------------------------

extern "C" void kernel_launch(void* const* d_in, const int* in_sizes, int n_in,
                              void* d_out, int out_size, void* d_ws, size_t ws_size,
                              hipStream_t stream) {
  (void)in_sizes; (void)n_in; (void)out_size; (void)ws_size;

  const float* x     = (const float*)d_in[0];
  const float* ln1_w = (const float*)d_in[1];
  const float* ln1_b = (const float*)d_in[2];
  const float* in_proj_w[2]  = {(const float*)d_in[3],  (const float*)d_in[11]};
  const float* conv_w[2]     = {(const float*)d_in[4],  (const float*)d_in[12]};
  const float* conv_b[2]     = {(const float*)d_in[5],  (const float*)d_in[13]};
  const float* dt_bias[2]    = {(const float*)d_in[6],  (const float*)d_in[14]};
  const float* A_log[2]      = {(const float*)d_in[7],  (const float*)d_in[15]};
  const float* Dp[2]         = {(const float*)d_in[8],  (const float*)d_in[16]};
  const float* norm_w[2]     = {(const float*)d_in[9],  (const float*)d_in[17]};
  const float* out_proj_w[2] = {(const float*)d_in[10], (const float*)d_in[18]};
  const float* proj_w = (const float*)d_in[19];
  const float* proj_b = (const float*)d_in[20];
  const float* ln2_w  = (const float*)d_in[21];
  const float* ln2_b  = (const float*)d_in[22];
  const float* fc1_w  = (const float*)d_in[23];
  const float* fc1_b  = (const float*)d_in[24];
  const float* fc2_w  = (const float*)d_in[25];
  const float* fc2_b  = (const float*)d_in[26];

  char* wsp = (char*)d_ws;
  auto take = [&](size_t bytes) -> void* {
    void* p = (void*)wsp;
    wsp += (bytes + 255) & ~(size_t)255;
    return p;
  };
  u16*   h_bf   = (u16*)  take((size_t)M_TOK * DIM * 2);
  u16*   zx_bf  = (u16*)  take((size_t)M_TOK * D_IN_PROJ * 2);
  float* xbc    = (float*)take((size_t)M_TOK * CONV_DIM * 4);
  float* ybuf   = (float*)take((size_t)M_TOK * D_INNER * 4);
  u16*   ynorm  = (u16*)  take((size_t)M_TOK * D_INNER * 2);
  u16*   concat = (u16*)  take((size_t)M_TOK * D_INNER * 2);
  float* x1     = (float*)take((size_t)M_TOK * DIM * 4);
  u16*   h2     = (u16*)  take((size_t)M_TOK * DIM * 2);
  u16*   mbuf   = (u16*)  take((size_t)M_TOK * MLP_HID * 2);
  u16* w_inp[2]  = {(u16*)take((size_t)D_IN_PROJ * DIM * 2),
                    (u16*)take((size_t)D_IN_PROJ * DIM * 2)};
  u16* w_outp[2] = {(u16*)take((size_t)DIM * D_INNER * 2),
                    (u16*)take((size_t)DIM * D_INNER * 2)};
  u16* w_proj = (u16*)take((size_t)DIM * D_INNER * 2);
  u16* w_fc1  = (u16*)take((size_t)MLP_HID * DIM * 2);
  u16* w_fc2  = (u16*)take((size_t)DIM * MLP_HID * 2);

  auto cvt = [&](const float* src, u16* dst, int n) {
    cvt_bf16<<<(n + 255) / 256, 256, 0, stream>>>(src, dst, n);
  };
  cvt(in_proj_w[0], w_inp[0], D_IN_PROJ * DIM);
  cvt(in_proj_w[1], w_inp[1], D_IN_PROJ * DIM);
  cvt(out_proj_w[0], w_outp[0], DIM * D_INNER);
  cvt(out_proj_w[1], w_outp[1], DIM * D_INNER);
  cvt(proj_w, w_proj, DIM * D_INNER);
  cvt(fc1_w, w_fc1, MLP_HID * DIM);
  cvt(fc2_w, w_fc2, DIM * MLP_HID);

  // LN1 -> bf16 activations (shared by fwd and bwd: in_proj commutes with flip)
  layernorm_to_bf16<<<M_TOK, 256, 0, stream>>>(x, ln1_w, ln1_b, h_bf);

  for (int dir = 0; dir < 2; ++dir) {
    // in_proj: (4096x1024)x(4160x1024)^T -> zxbcdt (bf16)
    gemm_wmma_bf16<true, false><<<dim3((D_IN_PROJ + 127) / 128, M_TOK / 128), 256, 0, stream>>>(
        h_bf, w_inp[dir], nullptr, zx_bf, nullptr, nullptr, D_IN_PROJ, DIM, D_IN_PROJ, 0);
    conv_silu<<<((M_TOK * CONV_DIM) + 255) / 256, 256, 0, stream>>>(
        zx_bf, conv_w[dir], conv_b[dir], xbc, dir);
    ssm_scan<<<B_SZ * NHEADS, 256, 0, stream>>>(
        zx_bf, xbc, dt_bias[dir], A_log[dir], Dp[dir], ybuf, dir);
    gated_rmsnorm<<<M_TOK, 256, 0, stream>>>(ybuf, zx_bf, norm_w[dir], ynorm, dir);
    // out_proj: writes bf16 directly into its half of the concat buffer (ldc=2048)
    gemm_wmma_bf16<true, false><<<dim3(DIM / 128, M_TOK / 128), 256, 0, stream>>>(
        ynorm, w_outp[dir], nullptr, concat, nullptr, nullptr, DIM, D_INNER, D_INNER, dir * DIM);
  }

  // proj + bias + residual(x) -> x1 (fp32)
  gemm_wmma_bf16<false, false><<<dim3(DIM / 128, M_TOK / 128), 256, 0, stream>>>(
      concat, w_proj, x1, nullptr, proj_b, x, DIM, D_INNER, DIM, 0);

  layernorm_to_bf16<<<M_TOK, 256, 0, stream>>>(x1, ln2_w, ln2_b, h2);

  // fc1 + bias + exact GELU -> bf16
  gemm_wmma_bf16<true, true><<<dim3(MLP_HID / 128, M_TOK / 128), 256, 0, stream>>>(
      h2, w_fc1, nullptr, mbuf, fc1_b, nullptr, MLP_HID, DIM, MLP_HID, 0);

  // fc2 + bias + residual(x1) -> d_out (fp32)
  gemm_wmma_bf16<false, false><<<dim3(DIM / 128, M_TOK / 128), 256, 0, stream>>>(
      mbuf, w_fc2, (float*)d_out, nullptr, fc2_b, x1, DIM, MLP_HID, DIM, 0);
}
